// KeywordSpottingModel_5248450036108
// MI455X (gfx1250) — matrix operand order repaired and checked by reading, not compiled, not run on hardware
//
#include <hip/hip_runtime.h>
#include <hip/hip_bf16.h>

// ---------------------------------------------------------------------------
// Mamba keyword-spotting forward for MI455X (gfx1250, wave32, WMMA).
//   pack_x -> pack_w (x3) -> WMMA gemm1 -> WMMA gemm2 -> conv+silu
//   -> WMMA gemm3 -> fused selective scan (+mean over T) -> tiny head
// GEMMs: v_wmma_f32_16x16x32_f16, 1 M-tile x NB N-tiles per wave so each
// A fragment feeds NB back-to-back WMMAs (NB=4 amortizes A traffic 4x).
// ---------------------------------------------------------------------------

typedef __attribute__((ext_vector_type(16))) _Float16 v16h;
typedef __attribute__((ext_vector_type(8)))  _Float16 v8h;
typedef __attribute__((ext_vector_type(8)))  float    v8f;

#define B_SZ   8
#define C_IN   80
#define T_LEN  1000
#define DMODEL 256
#define DINNER 512
#define DSTATE 16
#define DCONV  4
#define DTRANK 16
#define NLAB   35
#define ROWS   (B_SZ * T_LEN)   // 8000, multiple of 16

// ------------------------- packing kernels ---------------------------------

// x: (B, C_IN, T) f32  ->  XhH: (ROWS, 96) f16 row-major, K zero-padded 80->96
__global__ void pack_x_kernel(const float* __restrict__ x, _Float16* __restrict__ dst) {
    int idx = blockIdx.x * blockDim.x + threadIdx.x;
    const int KP = 96;
    if (idx >= ROWS * KP) return;
    int row = idx / KP, c = idx % KP;
    int b = row / T_LEN, t = row % T_LEN;
    float v = (c < C_IN) ? x[((size_t)b * C_IN + c) * T_LEN + t] : 0.0f;
    dst[idx] = (_Float16)v;
}

// src: (N, K) f32 row-major weight  ->  packed CDNA5 WMMA B-fragments (f16).
// dst[(((kt*tilesN)+tn)*32 + lane)*16 + i]:
//   lane<16 : elem i = B[kt*32 + {0..7,16..23}[i]][tn*16 + lane]
//   lane>=16: elem i = B[kt*32 + {8..15,24..31}[i]][tn*16 + (lane-16)]
// Each lane's fragment is one contiguous 32-byte chunk; consecutive tn are
// adjacent 1KB blocks so NB fragments clause into a single load burst.
__global__ void pack_w_kernel(const float* __restrict__ src, _Float16* __restrict__ dst,
                              int N, int K, int Kpad) {
    int idx = blockIdx.x * blockDim.x + threadIdx.x;
    int tilesN = N >> 4;
    int total = (Kpad >> 5) * tilesN * 32 * 16;
    if (idx >= total) return;
    int i    = idx & 15;
    int lane = (idx >> 4) & 31;
    int tile = idx >> 9;               // kt*tilesN + tn
    int tn = tile % tilesN, kt = tile / tilesN;
    int n = tn * 16 + (lane & 15);
    int k = kt * 32 + ((lane < 16) ? 0 : 8) + i + ((i < 8) ? 0 : 8);
    float v = (k < K) ? src[(size_t)n * K + k] : 0.0f;
    dst[idx] = (_Float16)v;
}

// ------------------------- WMMA GEMM ---------------------------------------
// C(M,N) = A(M,Kpad) * Bpacked(Kpad,N) [+ bias].  Each wave owns one 16-row
// strip and NB consecutive 16-col tiles; A fragment loaded once per k-step
// (2 x b128), then NB WMMAs issued against contiguous packed-B fragments.
// A-fragment per ISA 16-bit 16x32 layout:
//   lane<16: rows m=lane, k={0..7,16..23}; lane>=16: k={8..15,24..31}.
template <int NB>
__global__ __launch_bounds__(256)
void gemm_wmma_f16(const _Float16* __restrict__ A, const _Float16* __restrict__ Bp,
                   float* __restrict__ Cf, _Float16* __restrict__ Ch,
                   const float* __restrict__ bias, int M, int N, int Kpad) {
    int lane  = threadIdx.x & 31;
    int wave  = threadIdx.x >> 5;
    int tilesN  = N >> 4;
    int groupsN = tilesN / NB;
    int groups  = (M >> 4) * groupsN;
    int gw     = blockIdx.x * (blockDim.x >> 5) + wave;
    int stride = gridDim.x * (blockDim.x >> 5);
    int ktiles = Kpad >> 5;
    int halfSel = (lane < 16) ? 0 : 1;

    for (int g = gw; g < groups; g += stride) {
        int tm  = g / groupsN;
        int tn0 = (g % groupsN) * NB;
        v8f acc[NB];
#pragma unroll
        for (int j = 0; j < NB; ++j) acc[j] = (v8f){};

        const _Float16* arow = A + (size_t)(tm * 16 + (lane & 15)) * Kpad + halfSel * 8;
        for (int kt = 0; kt < ktiles; ++kt) {
            v8h lo = *(const v8h*)(arow + kt * 32);
            v8h hi = *(const v8h*)(arow + kt * 32 + 16);
            v16h a;
#pragma unroll
            for (int i = 0; i < 8; ++i) { a[i] = lo[i]; a[i + 8] = hi[i]; }
            const _Float16* bbase =
                Bp + ((size_t)(kt * tilesN + tn0) * 32 + lane) * 16;
#pragma unroll
            for (int j = 0; j < NB; ++j) {
                v16h b = *(const v16h*)(bbase + (size_t)j * 512);
                // 8 args: (neg_a, A, neg_b, B, c_mod, C, reuse_a, reuse_b)
                acc[j] = __builtin_amdgcn_wmma_f32_16x16x32_f16(
                    false, a, false, b, (short)0, acc[j], false, false);
            }
        }
        int rbase = tm * 16 + halfSel * 8;
#pragma unroll
        for (int j = 0; j < NB; ++j) {
            int colg = (tn0 + j) * 16 + (lane & 15);
            float badd = bias ? bias[colg] : 0.0f;
#pragma unroll
            for (int r = 0; r < 8; ++r) {
                float v = acc[j][r] + badd;
                size_t off = (size_t)(rbase + r) * N + colg;
                if (Ch) Ch[off] = (_Float16)v; else Cf[off] = v;
            }
        }
    }
}

// ------------------------- depthwise causal conv + SiLU --------------------
// xz: (ROWS, 1024) f32 (cols 0..511 = xh pre-conv).  Writes f32 + f16 copies.
__global__ void conv_silu_kernel(const float* __restrict__ xz,
                                 const float* __restrict__ conv_w,
                                 const float* __restrict__ conv_b,
                                 float* __restrict__ Xc, _Float16* __restrict__ XcH) {
    int idx = blockIdx.x * blockDim.x + threadIdx.x;
    if (idx >= ROWS * DINNER) return;
    int row = idx >> 9, d = idx & (DINNER - 1);
    int b = row / T_LEN, t = row % T_LEN;
    float acc = conv_b[d];
#pragma unroll
    for (int j = 0; j < DCONV; ++j) {
        int tt = t + j - (DCONV - 1);
        if (tt >= 0)
            acc += conv_w[d * DCONV + j] * xz[((size_t)(b * T_LEN + tt)) * 1024 + d];
    }
    float s = acc / (1.0f + __expf(-acc));   // SiLU
    Xc[idx]  = s;
    XcH[idx] = (_Float16)s;
}

// ------------------------- fused selective scan ----------------------------
// One block per batch, one thread per inner channel.  Fuses dt-projection,
// softplus, exp(delta*A) recurrence, C-contraction, D-skip, silu(z) gate and
// the mean over T (so the output projection runs on 8 rows, not 8000).
__global__ __launch_bounds__(512)
void scan_kernel(const float* __restrict__ dbc, const float* __restrict__ Xc,
                 const float* __restrict__ xz, const float* __restrict__ dt_w,
                 const float* __restrict__ dt_b, const float* __restrict__ A_log,
                 const float* __restrict__ Dp, float* __restrict__ ybar) {
    __shared__ float srow[DTRANK + 2 * DSTATE];   // 48 floats: dt | B | C
    int b = blockIdx.x, d = threadIdx.x;
    float wdt[DTRANK], Ad[DSTATE], h[DSTATE];
#pragma unroll
    for (int s = 0; s < DTRANK; ++s) wdt[s] = dt_w[d * DTRANK + s];
#pragma unroll
    for (int s = 0; s < DSTATE; ++s) { Ad[s] = -__expf(A_log[d * DSTATE + s]); h[s] = 0.0f; }
    float bdt = dt_b[d], Dd = Dp[d], acc = 0.0f;
    size_t rb = (size_t)b * T_LEN;

    for (int t = 0; t < T_LEN; ++t) {
        if (d < 48) srow[d] = dbc[(rb + t) * 48 + d];
        __syncthreads();
        float draw = bdt;
#pragma unroll
        for (int i = 0; i < DTRANK; ++i) draw += srow[i] * wdt[i];
        float delta = (draw > 20.0f) ? draw : __logf(1.0f + __expf(draw));
        float xh = Xc[(rb + t) * DINNER + d];
        float z  = xz[(rb + t) * 1024 + DINNER + d];
        float dx = delta * xh;
        float y = 0.0f;
#pragma unroll
        for (int s = 0; s < DSTATE; ++s) {
            float dA = __expf(delta * Ad[s]);
            h[s] = dA * h[s] + dx * srow[DTRANK + s];
            y += h[s] * srow[DTRANK + DSTATE + s];
        }
        acc += (y + xh * Dd) * (z / (1.0f + __expf(-z)));
        __syncthreads();
    }
    ybar[b * DINNER + d] = acc * (1.0f / (float)T_LEN);
}

// ------------------------- head: ybar @ out_proj^T @ fc^T + fc_b -----------
__global__ __launch_bounds__(256)
void head_kernel(const float* __restrict__ ybar, const float* __restrict__ out_w,
                 const float* __restrict__ fc_w, const float* __restrict__ fc_b,
                 float* __restrict__ out) {
    __shared__ float pooled[B_SZ * DMODEL];       // 8 KB
    int m = threadIdx.x;                           // 0..255
    for (int b = 0; b < B_SZ; ++b) {
        float a = 0.0f;
        for (int dd = 0; dd < DINNER; ++dd)
            a += ybar[b * DINNER + dd] * out_w[(size_t)m * DINNER + dd];
        pooled[b * DMODEL + m] = a;
    }
    __syncthreads();
    for (int i = threadIdx.x; i < B_SZ * NLAB; i += blockDim.x) {
        int b = i / NLAB, l = i % NLAB;
        float a = fc_b[l];
        for (int mm = 0; mm < DMODEL; ++mm)
            a += pooled[b * DMODEL + mm] * fc_w[(size_t)l * DMODEL + mm];
        out[i] = a;
    }
}

// ------------------------- launcher ----------------------------------------

static inline size_t rnd(size_t x) { return (x + 255) & ~(size_t)255; }

extern "C" void kernel_launch(void* const* d_in, const int* in_sizes, int n_in,
                              void* d_out, int out_size, void* d_ws, size_t ws_size,
                              hipStream_t stream) {
    const float* x        = (const float*)d_in[0];
    const float* proj_w   = (const float*)d_in[1];
    const float* proj_b   = (const float*)d_in[2];
    const float* in_proj  = (const float*)d_in[3];
    const float* conv_w   = (const float*)d_in[4];
    const float* conv_b   = (const float*)d_in[5];
    const float* x_proj   = (const float*)d_in[6];
    const float* dt_w     = (const float*)d_in[7];
    const float* dt_b     = (const float*)d_in[8];
    const float* A_log    = (const float*)d_in[9];
    const float* Dvec     = (const float*)d_in[10];
    const float* out_w    = (const float*)d_in[11];
    const float* fc_w     = (const float*)d_in[12];
    const float* fc_b     = (const float*)d_in[13];
    float* out = (float*)d_out;

    // workspace layout (~65 MB total)
    char* ws = (char*)d_ws;  size_t o = 0;
    _Float16* XhH = (_Float16*)(ws + o); o += rnd((size_t)ROWS * 96 * 2);
    _Float16* WpH = (_Float16*)(ws + o); o += rnd((size_t)96  * DMODEL * 2);
    _Float16* Hh  = (_Float16*)(ws + o); o += rnd((size_t)ROWS * DMODEL * 2);
    _Float16* WiH = (_Float16*)(ws + o); o += rnd((size_t)DMODEL * 1024 * 2);
    float*    XZ  = (float*)   (ws + o); o += rnd((size_t)ROWS * 1024 * 4);
    float*    Xc  = (float*)   (ws + o); o += rnd((size_t)ROWS * DINNER * 4);
    _Float16* XcH = (_Float16*)(ws + o); o += rnd((size_t)ROWS * DINNER * 2);
    _Float16* WxH = (_Float16*)(ws + o); o += rnd((size_t)DINNER * 48 * 2);
    float*    dbc = (float*)   (ws + o); o += rnd((size_t)ROWS * 48 * 4);
    float*    ybr = (float*)   (ws + o); o += rnd((size_t)B_SZ * DINNER * 4);
    (void)ws_size; (void)n_in; (void)in_sizes; (void)out_size;

    // 1) pack activations + weights into WMMA fragment layouts
    {
        int n = ROWS * 96;
        pack_x_kernel<<<(n + 255) / 256, 256, 0, stream>>>(x, XhH);
    }
    {
        int n = (96 / 32) * (DMODEL / 16) * 512;
        pack_w_kernel<<<(n + 255) / 256, 256, 0, stream>>>(proj_w, WpH, DMODEL, C_IN, 96);
        n = (DMODEL / 32) * (1024 / 16) * 512;
        pack_w_kernel<<<(n + 255) / 256, 256, 0, stream>>>(in_proj, WiH, 1024, DMODEL, DMODEL);
        n = (DINNER / 32) * (48 / 16) * 512;
        pack_w_kernel<<<(n + 255) / 256, 256, 0, stream>>>(x_proj, WxH, 48, DINNER, DINNER);
    }
    // 2) h = X @ proj_w^T + proj_b  (f16 out): 500 x (16/4) groups
    {
        int groups = (ROWS / 16) * (DMODEL / 16 / 4);
        gemm_wmma_f16<4><<<(groups + 7) / 8, 256, 0, stream>>>(XhH, WpH, nullptr, Hh,
                                                               proj_b, ROWS, DMODEL, 96);
    }
    // 3) xz = h @ in_proj_w^T  (f32 out): 500 x (64/4) groups
    {
        int groups = (ROWS / 16) * (1024 / 16 / 4);
        gemm_wmma_f16<4><<<(groups + 7) / 8, 256, 0, stream>>>(Hh, WiH, XZ, nullptr,
                                                               nullptr, ROWS, 1024, DMODEL);
    }
    // 4) causal depthwise conv + SiLU
    {
        int n = ROWS * DINNER;
        conv_silu_kernel<<<(n + 255) / 256, 256, 0, stream>>>(XZ, conv_w, conv_b, Xc, XcH);
    }
    // 5) dbc = xh @ x_proj_w^T: 500 x (3/3) groups
    {
        int groups = (ROWS / 16) * 1;
        gemm_wmma_f16<3><<<(groups + 7) / 8, 256, 0, stream>>>(XcH, WxH, dbc, nullptr,
                                                               nullptr, ROWS, 48, DINNER);
    }
    // 6) fused selective scan + time-mean
    scan_kernel<<<B_SZ, DINNER, 0, stream>>>(dbc, Xc, XZ, dt_w, dt_b, A_log, Dvec, ybr);
    // 7) head
    head_kernel<<<1, 256, 0, stream>>>(ybr, out_w, fc_w, fc_b, out);
}
